// CPAB_CL_53249004536360
// MI455X (gfx1250) — matrix operand order repaired and checked by reading, not compiled, not run on hardware
//
#include <hip/hip_runtime.h>

typedef _Float16 v16h __attribute__((ext_vector_type(16)));
typedef _Float16 v8h  __attribute__((ext_vector_type(8)));
typedef float    v8f  __attribute__((ext_vector_type(8)));

// ---------------------------------------------------------------------------
// CDNA5 async global->LDS copy (bypasses VGPRs, tracked by ASYNCcnt).
// Per-lane: copies 16 bytes from the lane's global address to the lane's LDS
// byte address.  Generic shared pointers carry the LDS byte offset in their
// low 32 bits on AMDGPU, so we pass that directly as the VDST operand.
// ---------------------------------------------------------------------------
__device__ __forceinline__ void async_lds_copy_b128(unsigned lds_off, const void* g)
{
    asm volatile("global_load_async_to_lds_b128 %0, %1, off"
                 :: "v"(lds_off), "v"(g) : "memory");
}
__device__ __forceinline__ void wait_async0()
{
    asm volatile("s_wait_asynccnt 0" ::: "memory");
}
// Issue one block-wide 32KB weight-matrix copy (256 threads x 8 x b128).
__device__ __forceinline__ void issue_weight_copy(_Float16* dst_lds,
                                                  const _Float16* src, int tid)
{
    unsigned base = (unsigned)(unsigned long long)(void*)dst_lds;
    const char* g = (const char*)src;
    #pragma unroll
    for (int i = 0; i < 8; ++i) {
        int byteoff = (tid + 256 * i) * 16;
        async_lds_copy_b128(base + byteoff, g + byteoff);
    }
}

// ---------------------------------------------------------------------------
// Prep kernels: build f16, column-major ([n][k]) weight copies in workspace,
// and fold  Wo @ blockdiag(B^T)  into a single 128x128 matrix Wc.
// ---------------------------------------------------------------------------

__global__ void prep_wt(const float* __restrict__ W1, const float* __restrict__ W2,
                        const float* __restrict__ W3,
                        _Float16* __restrict__ wt1, _Float16* __restrict__ wt2,
                        _Float16* __restrict__ wt3)
{
    int i = blockIdx.x * blockDim.x + threadIdx.x;   // 0 .. 3*16384-1
    int m  = i >> 14;
    int j  = i & 16383;
    int nn = j >> 7;          // output column
    int k  = j & 127;         // input row
    const float* W = (m == 0) ? W1 : (m == 1) ? W2 : W3;
    _Float16*    O = (m == 0) ? wt1 : (m == 1) ? wt2 : wt3;
    O[nn * 128 + k] = (_Float16)W[k * 128 + nn];     // transpose: [n][k]
}

__global__ void prep_wc(const float* __restrict__ Wo, const float* __restrict__ B,
                        _Float16* __restrict__ wct)
{
    int i  = blockIdx.x * blockDim.x + threadIdx.x;  // 0 .. 16383
    int nn = i >> 7;          // combined column u*32+j
    int k  = i & 127;
    int u  = nn >> 5;
    int jj = nn & 31;
    float s = 0.f;
    #pragma unroll
    for (int m = 0; m < 15; ++m)
        s += Wo[k * 60 + u * 15 + m] * B[jj * 15 + m];
    wct[nn * 128 + k] = (_Float16)s;                 // transposed: [n][k]
}

__global__ void prep_bc(const float* __restrict__ bo, const float* __restrict__ B,
                        float* __restrict__ bc)
{
    int nn = threadIdx.x;     // 0..127
    int u  = nn >> 5;
    int jj = nn & 31;
    float s = 0.f;
    #pragma unroll
    for (int m = 0; m < 15; ++m)
        s += bo[u * 15 + m] * B[jj * 15 + m];
    bc[nn] = s;
}

// ---------------------------------------------------------------------------
// Fused kernel: MLP (WMMA, async double-buffered weights) + combined
// output/basis GEMM + CPAB integration.
// Block = 256 threads (8 wave32), covers 128 batch rows.
// ---------------------------------------------------------------------------

__global__ __launch_bounds__(256)
void cpab_fused(const float* __restrict__ x,
                const float* __restrict__ W0, const float* __restrict__ b0,
                const float* __restrict__ b1, const float* __restrict__ b2,
                const float* __restrict__ b3,
                const _Float16* __restrict__ wt1, const _Float16* __restrict__ wt2,
                const _Float16* __restrict__ wt3, const _Float16* __restrict__ wct,
                const float* __restrict__ bc,
                float* __restrict__ zout, float* __restrict__ lout, int nrows)
{
    // LDS: xs(4K) | biases(2.5K) | W0(2K) | wbufA(32K) | wbufB(32K) | act(32K)
    // CPAB phase reuses wbufA+wbufB (64K contiguous) as A_full f32.
    __shared__ __align__(16) unsigned char smem[4096 + 2560 + 2048 + 3 * 32768];
    float*         xs    = (float*)smem;                 // 128 rows x 8 f32
    float*         biasL = (float*)(smem + 4096);        // b0,b1,b2,b3,bc : 5x128 f32
    float*         w0l   = (float*)(smem + 4096 + 2560); // 4x128 f32
    unsigned char* big   = smem + 8704;
    _Float16*      bufA  = (_Float16*)big;               // weights ping
    _Float16*      bufB  = (_Float16*)(big + 32768);     // weights pong
    _Float16*      actb  = (_Float16*)(big + 65536);     // 8 waves x 16x128 f16 [m][k]
    float*         Af    = (float*)big;                  // 128x128 f32 (reuse)

    const int tid  = threadIdx.x;
    const int lane = tid & 31;
    const int wave = tid >> 5;
    const int hi   = lane >> 4;     // lane group (C-layout M offset / frag K offset)
    const int ln   = lane & 15;

    const _Float16* wts[4] = {wt1, wt2, wt3, wct};
    _Float16*       wbufs[2] = {bufA, bufB};

    // kick off layer-0 weights immediately (async, overlaps all staging below)
    issue_weight_copy(bufA, wts[0], tid);

    // ---- stage x tile (coalesced b128), biases, W0 ----
    {
        float4 v = make_float4(0.f, 0.f, 0.f, 0.f);
        long gi = (long)blockIdx.x * 1024 + tid * 4;
        if (gi + 3 < (long)nrows * 8) v = *(const float4*)(x + gi);
        ((float4*)xs)[tid] = v;
    }
    for (int i = tid; i < 640; i += 256) {
        float v;
        if      (i < 128) v = b0[i];
        else if (i < 256) v = b1[i - 128];
        else if (i < 384) v = b2[i - 256];
        else if (i < 512) v = b3[i - 384];
        else              v = bc[i - 512];
        biasL[i] = v;
    }
    for (int i = tid; i < 512; i += 256) w0l[i] = W0[i];
    __syncthreads();

    // acc[t][v] holds C-layout f32 tile: n = 16t + ln, m = v + 8*hi (per wave)
    v8f acc[8];

    // ---- layer 0: h0 = x1 @ W0 + b0  (K=4, scalar VALU) ----
    #pragma unroll
    for (int v = 0; v < 8; ++v) {
        int mloc = wave * 16 + v + 8 * hi;
        float a0 = xs[mloc * 8 + 0], a1 = xs[mloc * 8 + 1];
        float a2 = xs[mloc * 8 + 2], a3 = xs[mloc * 8 + 3];
        #pragma unroll
        for (int t = 0; t < 8; ++t) {
            int nn = t * 16 + ln;
            acc[t][v] = biasL[nn] + a0 * w0l[nn] + a1 * w0l[128 + nn]
                                  + a2 * w0l[256 + nn] + a3 * w0l[384 + nn];
        }
    }

    _Float16* myact = actb + wave * 16 * 128;

    // ---- 4 GEMM layers: W1,W2,W3 (ReLU) and Wc (no ReLU) ----
    #pragma unroll
    for (int l = 0; l < 4; ++l) {
        wait_async0();        // my slice of wbufs[l&1] has landed in LDS
        __syncthreads();      // everyone's slice landed; prior reads of the
                              // other buffer are finished -> safe to overwrite
        if (l < 3) issue_weight_copy(wbufs[(l + 1) & 1], wts[l + 1], tid);
        const _Float16* wbuf = wbufs[l & 1];

        // spill activations f32->f16 into per-wave LDS slab, [m][k] layout
        #pragma unroll
        for (int t = 0; t < 8; ++t) {
            int nn = t * 16 + ln;
            #pragma unroll
            for (int v = 0; v < 8; ++v)
                myact[(v + 8 * hi) * 128 + nn] = (_Float16)acc[t][v];
        }
        __syncthreads();

        // init accumulators with bias
        #pragma unroll
        for (int t = 0; t < 8; ++t) {
            float bb = biasL[(l + 1) * 128 + t * 16 + ln];
            #pragma unroll
            for (int v = 0; v < 8; ++v) acc[t][v] = bb;
        }
        // K = 128 in 4 chunks of 32; 8 N-tiles -> 32 WMMA per layer
        #pragma unroll
        for (int kk = 0; kk < 4; ++kk) {
            const _Float16* ap = myact + ln * 128 + kk * 32 + hi * 8;
            v8h alo = *(const v8h*)ap;
            v8h ahi = *(const v8h*)(ap + 16);
            v16h af;
            #pragma unroll
            for (int i = 0; i < 8; ++i) { af[i] = alo[i]; af[8 + i] = ahi[i]; }
            #pragma unroll
            for (int t = 0; t < 8; ++t) {
                const _Float16* bp = wbuf + (t * 16 + ln) * 128 + kk * 32 + hi * 8;
                v8h blo = *(const v8h*)bp;
                v8h bhi = *(const v8h*)(bp + 16);
                v16h bf;
                #pragma unroll
                for (int i = 0; i < 8; ++i) { bf[i] = blo[i]; bf[8 + i] = bhi[i]; }
                acc[t] = __builtin_amdgcn_wmma_f32_16x16x32_f16(
                    false, af, false, bf, (short)0, acc[t], false, false);
            }
        }
        if (l < 3) {
            #pragma unroll
            for (int t = 0; t < 8; ++t)
                #pragma unroll
                for (int v = 0; v < 8; ++v) acc[t][v] = fmaxf(acc[t][v], 0.f);
        }
    }

    // ---- dump A_full (128 x [u*32+j]) to LDS (reuses weight buffers) ----
    __syncthreads();
    #pragma unroll
    for (int t = 0; t < 8; ++t) {
        int nn = t * 16 + ln;
        #pragma unroll
        for (int v = 0; v < 8; ++v)
            Af[(wave * 16 + v + 8 * hi) * 128 + nn] = acc[t][v];
    }
    __syncthreads();

    // ---- CPAB closed-form integration: 512 elements / block, 2 per thread ----
    const float eps = 1e-8f;
    for (int e = tid; e < 512; e += 256) {
        int row = e >> 2, u = e & 3;
        const float* Ar = Af + row * 128 + u * 32;   // [a0,b0,a1,b1,...]
        float xx = xs[row * 8 + 4 + u];
        float tt = 1.f, logg = 0.f;
        int c = (int)floorf(xx * 16.f);
        c = c < 0 ? 0 : (c > 15 ? 15 : c);
        bool done = false;
        for (int it = 0; it < 18; ++it) {
            float a = Ar[2 * c], b = Ar[2 * c + 1];
            float v = a * xx + b;
            bool small_a = fabsf(a) < eps;
            float a_safe = small_a ? 1.f : a;
            float eat = expf(a * tt);
            float psi = small_a ? (xx + b * tt)
                                : (eat * xx + (b / a_safe) * (eat - 1.f));
            float left  = (float)c * (1.f / 16.f);
            float right = (float)(c + 1) * (1.f / 16.f);
            bool inside = (psi >= left) && (psi <= right);
            float xc = (v >= 0.f) ? right : left;
            float v_safe = (fabsf(v) < eps) ? 1.f : v;
            float b_safe = (fabsf(b) < eps) ? 1.f : b;
            float ratio = fmaxf((a * xc + b) / v_safe, eps);
            float thit = small_a ? ((xc - xx) / b_safe) : (logf(ratio) / a_safe);
            float tau  = inside ? tt : thit;
            bool nd = done || inside;
            xx   = done ? xx   : (inside ? psi : xc);
            logg = done ? logg : (logg + a * tau);
            tt   = nd ? tt : (tt - thit);
            int cs = c + ((v >= 0.f) ? 1 : -1);
            cs = cs < 0 ? 0 : (cs > 15 ? 15 : cs);
            c = nd ? c : cs;
            done = nd;
        }
        long g = (long)blockIdx.x * 128 + row;
        if (g < nrows) {
            zout[g * 8 + 4 + u] = xx;
            lout[g * 8 + 4 + u] = logg;
        }
    }
    // lower half: z = x1, log-grad = 0
    for (int i = tid; i < 512; i += 256) {
        int row = i >> 2, col = i & 3;
        long g = (long)blockIdx.x * 128 + row;
        if (g < nrows) {
            zout[g * 8 + col] = xs[row * 8 + col];
            lout[g * 8 + col] = 0.f;
        }
    }
}

// ---------------------------------------------------------------------------

extern "C" void kernel_launch(void* const* d_in, const int* in_sizes, int n_in,
                              void* d_out, int out_size, void* d_ws, size_t ws_size,
                              hipStream_t stream)
{
    const float* x  = (const float*)d_in[0];
    const float* W0 = (const float*)d_in[1];
    const float* b0 = (const float*)d_in[2];
    const float* W1 = (const float*)d_in[3];
    const float* b1 = (const float*)d_in[4];
    const float* W2 = (const float*)d_in[5];
    const float* b2 = (const float*)d_in[6];
    const float* W3 = (const float*)d_in[7];
    const float* b3 = (const float*)d_in[8];
    const float* Wo = (const float*)d_in[9];
    const float* bo = (const float*)d_in[10];
    const float* B  = (const float*)d_in[11];
    int nrows = in_sizes[0] / 8;

    // workspace: 4 x 32KB f16 weights + 512B bc  (131584 bytes)
    _Float16* wt1 = (_Float16*)d_ws;
    _Float16* wt2 = wt1 + 16384;
    _Float16* wt3 = wt2 + 16384;
    _Float16* wct = wt3 + 16384;
    float*    bcb = (float*)(wct + 16384);

    prep_wt<<<192, 256, 0, stream>>>(W1, W2, W3, wt1, wt2, wt3);
    prep_wc<<<64, 256, 0, stream>>>(Wo, B, wct);
    prep_bc<<<1, 128, 0, stream>>>(bo, B, bcb);

    float* zout = (float*)d_out;
    float* lout = zout + (long)nrows * 8;
    int blocks = (nrows + 127) / 128;
    cpab_fused<<<blocks, 256, 0, stream>>>(x, W0, b0, b1, b2, b3,
                                           wt1, wt2, wt3, wct, bcb,
                                           zout, lout, nrows);
}
